// SurfaceToVolumeCVAE_60679297958330
// MI455X (gfx1250) — compile-verified, matrix operand order checked by reading, not run on hardware
//
#include <hip/hip_runtime.h>
#include <hip/hip_bf16.h>

typedef __attribute__((ext_vector_type(16))) _Float16 v16h;
typedef __attribute__((ext_vector_type(8)))  float    v8f;
typedef __attribute__((ext_vector_type(4)))  unsigned u32x4;

union FragH { v16h h; unsigned int u[8]; };

#define BB   4
#define NPT  2048
#define KNN  20
#define BNPT 8192            // BB*NPT
#define MKNN 163840          // BNPT*KNN
#define NINT 4096
#define MFOLD 16384          // BB*NINT
#define FIN  1568            // round32(1539)
#define LDS_STRIDE 40        // 32 K-halfs + 8 pad

// ---------------- order-preserving float<->u32 (for atomic max) -------------
__device__ __forceinline__ unsigned int enc_ord(float f) {
  unsigned int b = __float_as_uint(f);
  return (b & 0x80000000u) ? ~b : (b | 0x80000000u);
}
__device__ __forceinline__ float dec_ord(unsigned int u) {
  unsigned int b = (u & 0x80000000u) ? (u & 0x7FFFFFFFu) : ~u;
  return __uint_as_float(b);
}

// low 32 bits of a generic LDS pointer == LDS byte offset (aperture mapping)
__device__ __forceinline__ unsigned lds_addr32(const void* p) {
  return (unsigned)(unsigned long long)p;
}

// gfx1250 async copy: LDS[lds_byte] <= MEM[g], 16B per lane (ASYNCcnt)
__device__ __forceinline__ void async_ld_b128(unsigned lds_byte, const void* g) {
  asm volatile("global_load_async_to_lds_b128 %0, %1, off"
               :: "v"(lds_byte), "v"(g) : "memory");
}
__device__ __forceinline__ void wait_async_2() {
  asm volatile("s_wait_asynccnt 2" ::: "memory");
}
__device__ __forceinline__ void wait_async_0() {
  asm volatile("s_wait_asynccnt 0" ::: "memory");
}

// one 16x16x32 step for this wave's two output tiles, operands in LDS
__device__ __forceinline__ void frag_mma(const _Float16* sa, const _Float16* sb,
                                         int tm, int tn, int half, int ln16,
                                         v8f& acc0, v8f& acc1) {
  FragH fa, fb0, fb1;
#pragma unroll
  for (int q = 0; q < 8; ++q) {
    int kk = (q & 3) * 2 + (q >> 2) * 16 + half * 8;   // ISA 7.12.2 16-bit layout
    fa.u[q]  = *(const unsigned int*)(sa + (tm * 16 + ln16) * LDS_STRIDE + kk);
    fb0.u[q] = *(const unsigned int*)(sb + (tn + ln16) * LDS_STRIDE + kk);
    fb1.u[q] = *(const unsigned int*)(sb + (tn + 16 + ln16) * LDS_STRIDE + kk);
  }
  acc0 = __builtin_amdgcn_wmma_f32_16x16x32_f16(false, fa.h, false, fb0.h,
                                                (short)0, acc0, false, false);
  acc1 = __builtin_amdgcn_wmma_f32_16x16x32_f16(false, fa.h, false, fb1.h,
                                                (short)0, acc1, false, false);
}

// ---------------- generic WMMA GEMM: C[M,N] = A[M,Kp] * B[N,Kp]^T -----------
// A,B f16 row-major, stride Kp (Kp % 32 == 0, zero padded).
// epi: 0=none  1=+bias  2=+bias,relu  3=*scale+shift,lrelu
//      4=*scale+shift,lrelu, then atomicMax into Cmax[(m/grp)*ldcmax + n]
__global__ void __launch_bounds__(256)
wmma_gemm(const _Float16* __restrict__ A, long long sAz,
          const _Float16* __restrict__ Bm, long long sBz, int Kp,
          float* __restrict__ C, long long sCz, int ldc,
          _Float16* __restrict__ C16, int ldc16,
          unsigned int* __restrict__ Cmax, int ldcmax, int grp,
          const float* __restrict__ p0, const float* __restrict__ p1,
          int M, int Nn, int epi)
{
  __shared__ _Float16 sA[2][64 * LDS_STRIDE];
  __shared__ _Float16 sB[2][64 * LDS_STRIDE];

  A  += (long long)blockIdx.z * sAz;
  Bm += (long long)blockIdx.z * sBz;
  if (C) C += (long long)blockIdx.z * sCz;

  const int t    = threadIdx.x;
  const int wave = t >> 5;
  const int lane = t & 31;
  const int half = lane >> 4;
  const int ln16 = lane & 15;
  const int tm   = wave >> 1;       // m-subtile 0..3
  const int tn   = (wave & 1) * 32; // n-subtile base

  const int m0 = blockIdx.x * 64;
  const int n0 = blockIdx.y * 64;

  // staging role: one 16B chunk per thread (row 0..63, seg 0..3)
  const int row  = t >> 2;
  const int segh = (t & 3) * 8;     // f16 column of this thread's chunk

  v8f acc0 = {};
  v8f acc1 = {};

  const int nStage = Kp >> 5;
  const bool interior = (m0 + 64 <= M) && (n0 + 64 <= Nn);

  if (interior) {
    // ---- async double-buffered pipeline (gfx1250 ASYNCcnt path) ----
    const _Float16* gA = A  + (long long)(m0 + row) * Kp + segh;
    const _Float16* gB = Bm + (long long)(n0 + row) * Kp + segh;
    unsigned lA[2], lB[2];
    lA[0] = lds_addr32(&sA[0][row * LDS_STRIDE + segh]);
    lA[1] = lds_addr32(&sA[1][row * LDS_STRIDE + segh]);
    lB[0] = lds_addr32(&sB[0][row * LDS_STRIDE + segh]);
    lB[1] = lds_addr32(&sB[1][row * LDS_STRIDE + segh]);

    async_ld_b128(lA[0], gA);
    async_ld_b128(lB[0], gB);

    for (int s = 0; s < nStage; ++s) {
      const int buf = s & 1;
      if (s + 1 < nStage) {
        async_ld_b128(lA[buf ^ 1], gA + (s + 1) * 32);
        async_ld_b128(lB[buf ^ 1], gB + (s + 1) * 32);
        wait_async_2();                 // stage s (2 older ops) complete
      } else {
        wait_async_0();
      }
      __syncthreads();                  // buf visible to all waves
      frag_mma(sA[buf], sB[buf], tm, tn, half, ln16, acc0, acc1);
      __syncthreads();                  // safe to overwrite buf next issue
    }
  } else {
    // ---- guarded synchronous path (block touches M/N edge) ----
    const bool okA = (m0 + row) < M;
    const bool okB = (n0 + row) < Nn;
    const _Float16* gA = A  + (long long)(m0 + row) * Kp + segh;
    const _Float16* gB = Bm + (long long)(n0 + row) * Kp + segh;
    for (int s = 0; s < nStage; ++s) {
      u32x4 va = {}, vb = {};
      if (okA) va = *(const u32x4*)(gA + s * 32);
      if (okB) vb = *(const u32x4*)(gB + s * 32);
      *(u32x4*)(&sA[0][row * LDS_STRIDE + segh]) = va;
      *(u32x4*)(&sB[0][row * LDS_STRIDE + segh]) = vb;
      __syncthreads();
      frag_mma(sA[0], sB[0], tm, tn, half, ln16, acc0, acc1);
      __syncthreads();
    }
  }

  // epilogue: acc[r] at (m = r + half*8, n = ln16) within each 16x16 tile
#pragma unroll
  for (int r = 0; r < 8; ++r) {
    int gm = m0 + tm * 16 + half * 8 + r;
    if (gm >= M) continue;
#pragma unroll
    for (int c = 0; c < 2; ++c) {
      int gn = n0 + tn + c * 16 + ln16;
      if (gn >= Nn) continue;
      float v = (c == 0) ? acc0[r] : acc1[r];
      if (epi == 4) {
        float h = v * p0[gn] + p1[gn];
        h = h > 0.f ? h : 0.2f * h;
        atomicMax(&Cmax[(long long)(gm / grp) * ldcmax + gn], enc_ord(h));
        continue;
      }
      if (epi == 1)      { v += p0[gn]; }
      else if (epi == 2) { v += p0[gn]; v = v > 0.f ? v : 0.f; }
      else if (epi == 3) { v = v * p0[gn] + p1[gn]; v = v > 0.f ? v : 0.2f * v; }
      if (C)   C[(long long)gm * ldc + gn] = v;
      if (C16) C16[(long long)gm * ldc16 + gn] = (_Float16)v;
    }
  }
}

// ---------------- small helper kernels --------------------------------------
__global__ void bnprep_kernel(const float* __restrict__ g, const float* __restrict__ b,
                              const float* __restrict__ m, const float* __restrict__ v,
                              float* __restrict__ scale, float* __restrict__ shift, int C) {
  int i = blockIdx.x * blockDim.x + threadIdx.x;
  if (i >= C) return;
  float s = g[i] * rsqrtf(v[i] + 1e-5f);
  scale[i] = s;
  shift[i] = b[i] - m[i] * s;
}

__global__ void cvt_pad_kernel(const float* __restrict__ src, _Float16* __restrict__ dst,
                               int K, int Kp, long long total) {
  long long t = (long long)blockIdx.x * blockDim.x + threadIdx.x;
  if (t >= total) return;
  int c = (int)(t % Kp);
  long long r = t / Kp;
  dst[t] = (c < K) ? (_Float16)src[r * K + c] : (_Float16)0.f;
}

__global__ void diag_kernel(const float* __restrict__ gram, float* __restrict__ dg) {
  int t = blockIdx.x * blockDim.x + threadIdx.x;
  if (t >= BNPT) return;
  int b = t >> 11, m = t & 2047;
  dg[t] = gram[((long long)b * NPT + m) * NPT + m];
}

__global__ void knn_kernel(const float* __restrict__ gram, const float* __restrict__ dg,
                           int* __restrict__ idx) {
  int t = blockIdx.x * blockDim.x + threadIdx.x;
  if (t >= BNPT) return;
  int b = t >> 11, n = t & 2047;
  const float* row = gram + (long long)t * NPT;
  const float* d0  = dg + (b << 11);
  float snn = d0[n];
  float bd[KNN]; int bi[KNN];
#pragma unroll
  for (int i = 0; i < KNN; ++i) { bd[i] = 3.4e38f; bi[i] = 0; }
  for (int m = 0; m < NPT; ++m) {
    float d = snn + d0[m] - 2.f * row[m];
    if (d < bd[KNN - 1]) {
      int j = KNN - 1;
      while (j > 0 && bd[j - 1] > d) { bd[j] = bd[j - 1]; bi[j] = bi[j - 1]; --j; }
      bd[j] = d; bi[j] = m;
    }
  }
#pragma unroll
  for (int k = 0; k < KNN; ++k) idx[t * KNN + k] = bi[k];
}

// ef[(b,n,k), c] : c<D -> x[nbr]-x[ctr] ; D<=c<2D -> x[ctr] ; pad 0
__global__ void edgefeat_kernel(const _Float16* __restrict__ xh, int Dp, int D,
                                const int* __restrict__ idx, _Float16* __restrict__ ef,
                                int twoDp, long long total) {
  long long t = (long long)blockIdx.x * blockDim.x + threadIdx.x;
  if (t >= total) return;
  int c = (int)(t % twoDp);
  long long r  = t / twoDp;   // 0..MKNN-1
  long long pn = r / KNN;     // b*NPT + n
  int b = (int)(pn >> 11);
  _Float16 v = (_Float16)0.f;
  if (c < D) {
    int nb = idx[r];
    v = xh[((long long)(b << 11) + nb) * Dp + c] - xh[pn * Dp + c];
  } else if (c < 2 * D) {
    v = xh[pn * Dp + (c - D)];
  }
  ef[t] = v;
}

__global__ void maxdecode_kernel(const unsigned int* __restrict__ xmax, int O,
                                 _Float16* __restrict__ xnext,
                                 _Float16* __restrict__ catH, int catOff, long long total) {
  long long t = (long long)blockIdx.x * blockDim.x + threadIdx.x;
  if (t >= total) return;
  int c = (int)(t % O);
  long long pt = t / O;
  _Float16 h = (_Float16)dec_ord(xmax[t]);
  if (xnext) xnext[pt * O + c] = h;             // next layer Dp == O
  catH[pt * 512 + catOff + c] = h;
}

__global__ void pool_kernel(const float* __restrict__ H, float* __restrict__ featF,
                            _Float16* __restrict__ featH) {
  int b = blockIdx.x;
  int c = blockIdx.y * blockDim.x + threadIdx.x;   // 0..511
  const float* p = H + (long long)b * NPT * 512 + c;
  float mx = -3.4e38f, sm = 0.f;
  for (int n = 0; n < NPT; ++n) { float v = p[(long long)n * 512]; mx = fmaxf(mx, v); sm += v; }
  float mean = sm * (1.f / (float)NPT);
  featF[b * 1024 + c]       = mx;
  featF[b * 1024 + 512 + c] = mean;
  featH[b * 1024 + c]       = (_Float16)mx;
  featH[b * 1024 + 512 + c] = (_Float16)mean;
}

// inH[(b,n), c] : c<3 -> first3 ; c<515 -> mu[b] ; c<1539 -> feat[b] ; pad 0
__global__ void foldin_kernel(const float* __restrict__ first3, int tplMode,
                              const float* __restrict__ mu, const float* __restrict__ featF,
                              _Float16* __restrict__ inH, long long total) {
  long long t = (long long)blockIdx.x * blockDim.x + threadIdx.x;
  if (t >= total) return;
  int c = (int)(t % FIN);
  long long r = t / FIN;          // 0..MFOLD-1
  int b = (int)(r >> 12);
  int n = (int)(r & 4095);
  float v = 0.f;
  if (c < 3)          v = tplMode ? first3[n * 3 + c] : first3[r * 3 + c];
  else if (c < 515)   v = mu[b * 512 + (c - 3)];
  else if (c < 1539)  v = featF[b * 1024 + (c - 515)];
  inH[t] = (_Float16)v;
}

// ---------------- host side --------------------------------------------------
static inline void launch_gemm(hipStream_t s,
    const _Float16* A, long long sAz, const _Float16* B, long long sBz, int Kp,
    float* C, long long sCz, int ldc, _Float16* C16, int ldc16,
    unsigned int* Cmax, int ldcmax, int grp,
    const float* p0, const float* p1, int M, int Nn, int nz, int epi) {
  dim3 g((unsigned)((M + 63) / 64), (unsigned)((Nn + 63) / 64), (unsigned)nz);
  wmma_gemm<<<g, dim3(256), 0, s>>>(A, sAz, B, sBz, Kp, C, sCz, ldc, C16, ldc16,
                                    Cmax, ldcmax, grp, p0, p1, M, Nn, epi);
}

static inline unsigned gblk(long long total) { return (unsigned)((total + 255) / 256); }

extern "C" void kernel_launch(void* const* d_in, const int* in_sizes, int n_in,
                              void* d_out, int out_size, void* d_ws, size_t ws_size,
                              hipStream_t stream) {
  (void)in_sizes; (void)n_in; (void)out_size; (void)ws_size;
  auto F = [&](int i) { return (const float*)d_in[i]; };

  // input map (setup_inputs insertion order)
  const float* surf  = F(0);
  const float* w_ec[4] = {F(1), F(6), F(11), F(16)};
  const float* bn_g[5] = {F(2), F(7), F(12), F(17), F(22)};
  const float* bn_b[5] = {F(3), F(8), F(13), F(18), F(23)};
  const float* bn_m[5] = {F(4), F(9), F(14), F(19), F(24)};
  const float* bn_v[5] = {F(5), F(10), F(15), F(20), F(25)};
  const float* wa  = F(21);
  const float* wmu = F(26); const float* bmu = F(27);
  const float* wlv = F(28); const float* blv = F(29);
  const float* fw[2][4] = {{F(30), F(32), F(34), F(36)}, {F(38), F(40), F(42), F(44)}};
  const float* fb[2][4] = {{F(31), F(33), F(35), F(37)}, {F(39), F(41), F(43), F(45)}};
  const float* tpl = F(46);
  float* outF = (float*)d_out;          // [out2 49152][mu 2048][logvar 2048]

  // ---- workspace layout ----
  char* ws = (char*)d_ws;
  size_t off = 0;
  auto alloc = [&](size_t bytes) -> char* {
    off = (off + 255) & ~(size_t)255;
    char* p = ws + off; off += bytes; return p;
  };
  const int bnC[5] = {64, 64, 128, 256, 512};
  float* bnS[5]; float* bnSh[5];
  for (int i = 0; i < 5; ++i) { bnS[i] = (float*)alloc(bnC[i] * 4); bnSh[i] = (float*)alloc(bnC[i] * 4); }

  const int Din[4]   = {3, 64, 64, 128};
  const int Dpin[4]  = {32, 64, 64, 128};
  const int Oc[4]    = {64, 64, 128, 256};
  const int twoDp[4] = {32, 128, 128, 256};
  const int catOff[4]= {0, 64, 128, 256};

  _Float16* wEh[4];
  for (int l = 0; l < 4; ++l) wEh[l] = (_Float16*)alloc((size_t)Oc[l] * twoDp[l] * 2);
  _Float16* waH  = (_Float16*)alloc((size_t)512 * 512 * 2);
  _Float16* wmuH = (_Float16*)alloc((size_t)512 * 1024 * 2);
  _Float16* wlvH = (_Float16*)alloc((size_t)512 * 1024 * 2);
  const int fRows[4] = {512, 512, 256, 3};
  const int fK[4]    = {1539, 512, 512, 256};
  const int fKp[4]   = {FIN, 512, 512, 256};
  _Float16* fwH[2][4];
  for (int d = 0; d < 2; ++d)
    for (int l = 0; l < 4; ++l) fwH[d][l] = (_Float16*)alloc((size_t)fRows[l] * fKp[l] * 2);

  _Float16* xh[4];
  for (int l = 0; l < 4; ++l) xh[l] = (_Float16*)alloc((size_t)BNPT * Dpin[l] * 2);
  _Float16* catH = (_Float16*)alloc((size_t)BNPT * 512 * 2);
  int*   idx  = (int*)alloc((size_t)MKNN * 4);
  float* dg   = (float*)alloc((size_t)BNPT * 4);
  float* gram = (float*)alloc((size_t)BB * NPT * NPT * 4);
  _Float16* ef = (_Float16*)alloc((size_t)MKNN * 256 * 2);
  unsigned int* xmax = (unsigned int*)alloc((size_t)BNPT * 256 * 4);
  float* Hwa   = (float*)alloc((size_t)BNPT * 512 * 4);
  float* featF = (float*)alloc((size_t)BB * 1024 * 4);
  _Float16* featH = (_Float16*)alloc((size_t)BB * 1024 * 2);
  _Float16* inH = (_Float16*)alloc((size_t)MFOLD * FIN * 2);
  _Float16* hAh = (_Float16*)alloc((size_t)MFOLD * 512 * 2);
  _Float16* hBh = (_Float16*)alloc((size_t)MFOLD * 512 * 2);
  _Float16* hCh = (_Float16*)alloc((size_t)MFOLD * 256 * 2);
  float* out1 = (float*)alloc((size_t)MFOLD * 3 * 4);

  // ---- BN folding ----
  for (int i = 0; i < 5; ++i)
    bnprep_kernel<<<gblk(bnC[i]), 256, 0, stream>>>(bn_g[i], bn_b[i], bn_m[i], bn_v[i],
                                                    bnS[i], bnSh[i], bnC[i]);
  // ---- f32 -> f16 (K padded to %32) ----
  struct Cvt { const float* s; _Float16* d; int rows, K, Kp; };
  Cvt cv[32]; int nc = 0;
  cv[nc++] = {surf, xh[0], BNPT, 3, 32};
  for (int l = 0; l < 4; ++l) cv[nc++] = {w_ec[l], wEh[l], Oc[l], 2 * Din[l], twoDp[l]};
  cv[nc++] = {wa, waH, 512, 512, 512};
  cv[nc++] = {wmu, wmuH, 512, 1024, 1024};
  cv[nc++] = {wlv, wlvH, 512, 1024, 1024};
  for (int d = 0; d < 2; ++d)
    for (int l = 0; l < 4; ++l) cv[nc++] = {fw[d][l], fwH[d][l], fRows[l], fK[l], fKp[l]};
  for (int i = 0; i < nc; ++i) {
    long long tot = (long long)cv[i].rows * cv[i].Kp;
    cvt_pad_kernel<<<gblk(tot), 256, 0, stream>>>(cv[i].s, cv[i].d, cv[i].K, cv[i].Kp, tot);
  }

  // ---- DGCNN encoder: 4 edge convs ----
  for (int l = 0; l < 4; ++l) {
    // Gram S = X X^T per batch (WMMA)
    launch_gemm(stream, xh[l], (long long)NPT * Dpin[l], xh[l], (long long)NPT * Dpin[l],
                Dpin[l], gram, (long long)NPT * NPT, NPT, nullptr, 0,
                nullptr, 0, 1, nullptr, nullptr, NPT, NPT, BB, 0);
    diag_kernel<<<gblk(BNPT), 256, 0, stream>>>(gram, dg);
    knn_kernel<<<gblk(BNPT), 256, 0, stream>>>(gram, dg, idx);

    long long efTot = (long long)MKNN * twoDp[l];
    edgefeat_kernel<<<gblk(efTot), 256, 0, stream>>>(xh[l], Dpin[l], Din[l], idx, ef,
                                                     twoDp[l], efTot);
    hipMemsetAsync(xmax, 0, (size_t)BNPT * Oc[l] * 4, stream);
    // edge GEMM with fused BN + LeakyReLU + max-over-neighbors (atomic)
    launch_gemm(stream, ef, 0, wEh[l], 0, twoDp[l], nullptr, 0, 0, nullptr, 0,
                xmax, Oc[l], KNN, bnS[l], bnSh[l], MKNN, Oc[l], 1, 4);
    long long dTot = (long long)BNPT * Oc[l];
    maxdecode_kernel<<<gblk(dTot), 256, 0, stream>>>(xmax, Oc[l],
                                                     (l < 3) ? xh[l + 1] : nullptr,
                                                     catH, catOff[l], dTot);
  }

  // ---- aggregation conv (wa) + BN + lrelu, then max/mean pool ----
  launch_gemm(stream, catH, 0, waH, 0, 512, Hwa, 0, 512, nullptr, 0,
              nullptr, 0, 1, bnS[4], bnSh[4], BNPT, 512, 1, 3);
  pool_kernel<<<dim3(BB, 2), 256, 0, stream>>>(Hwa, featF, featH);

  // ---- VAE heads (z = mu) ----
  float* muOut = outF + (size_t)MFOLD * 3;         // 49152
  float* lvOut = muOut + BB * 512;                 // 51200
  launch_gemm(stream, featH, 0, wmuH, 0, 1024, muOut, 0, 512, nullptr, 0,
              nullptr, 0, 1, bmu, nullptr, BB, 512, 1, 1);
  launch_gemm(stream, featH, 0, wlvH, 0, 1024, lvOut, 0, 512, nullptr, 0,
              nullptr, 0, 1, blv, nullptr, BB, 512, 1, 1);

  // ---- folding decoder: two 4-layer MLPs ----
  long long inTot = (long long)MFOLD * FIN;
  for (int d = 0; d < 2; ++d) {
    foldin_kernel<<<gblk(inTot), 256, 0, stream>>>((d == 0) ? tpl : out1, (d == 0) ? 1 : 0,
                                                   muOut, featF, inH, inTot);
    launch_gemm(stream, inH, 0, fwH[d][0], 0, FIN, nullptr, 0, 0, hAh, 512,
                nullptr, 0, 1, fb[d][0], nullptr, MFOLD, 512, 1, 2);
    launch_gemm(stream, hAh, 0, fwH[d][1], 0, 512, nullptr, 0, 0, hBh, 512,
                nullptr, 0, 1, fb[d][1], nullptr, MFOLD, 512, 1, 2);
    launch_gemm(stream, hBh, 0, fwH[d][2], 0, 512, nullptr, 0, 0, hCh, 256,
                nullptr, 0, 1, fb[d][2], nullptr, MFOLD, 256, 1, 2);
    float* last = (d == 0) ? out1 : outF;          // final layer -> out2
    launch_gemm(stream, hCh, 0, fwH[d][3], 0, 256, last, 0, 3, nullptr, 0,
                nullptr, 0, 1, fb[d][3], nullptr, MFOLD, 3, 1, 1);
  }
}
